// AttentionAggregator_85899346241
// MI455X (gfx1250) — matrix operand order repaired and checked by reading, not compiled, not run on hardware
//
#include <hip/hip_runtime.h>
#include <hip/hip_bf16.h>

// ---------- types ----------
typedef __attribute__((ext_vector_type(16))) __bf16       v16bf;
typedef __attribute__((ext_vector_type(8)))  float        v8f;
typedef __attribute__((ext_vector_type(4)))  unsigned int u32x4;
typedef __attribute__((ext_vector_type(4)))  float        f32x4;

#define NHEADS 2
#define DIM    128      // D == T == Dout == 128
#define NS     32       // neighbors per node
#define AROWS  48       // 32 neighbors + 1 self + 15 zero pad (3 row tiles of 16)
#define NT_ROWS 34      // rows of neigh_t kept in LDS (0..32 used)
#define NT_STR 132      // padded f32 stride for neigh_t in LDS (bank-conflict dodge)
#define LEAKY  0.2f

struct Pair128 { u32x4 lo, hi; };

__device__ __forceinline__ unsigned short f32_to_bf16(float f) {
  unsigned int u = __float_as_uint(f);
  unsigned int r = 0x7FFFu + ((u >> 16) & 1u);   // round-to-nearest-even
  return (unsigned short)((u + r) >> 16);
}

// LDS byte offset of a generic pointer into shared memory (AMDGPU: low 32 bits).
__device__ __forceinline__ unsigned lds_off(const void* p) {
  return (unsigned)(uintptr_t)p;
}

// Async global->LDS 16B copy per lane (VGLOBAL GLOBAL_LOAD_ASYNC_TO_LDS_B128).
// Tracked by ASYNCcnt; no VGPR data round-trip.
__device__ __forceinline__ void async_g2l_b128(unsigned ldsDst, const void* gsrc) {
  asm volatile("global_load_async_to_lds_b128 %0, %1, off"
               :: "v"(ldsDst), "v"(gsrc) : "memory");
}
__device__ __forceinline__ void wait_async0() {
  asm volatile("s_wait_asynccnt 0x0" ::: "memory");
}

// A fragment: 16x32 bf16, row-major source with given stride (elements).
// lanes 0-15: row = base+lane, K = k0+0..7 | k0+16..23
// lanes 16-31: row = base+lane-16, K = k0+8..15 | k0+24..31
__device__ __forceinline__ v16bf frag_a(const unsigned short* base, int stride,
                                        int rowBase, int k0, int lane) {
  const unsigned short* p = base + (rowBase + (lane & 15)) * stride
                                 + k0 + ((lane >> 4) << 3);
  Pair128 t;
  t.lo = *reinterpret_cast<const u32x4*>(p);
  t.hi = *reinterpret_cast<const u32x4*>(p + 16);
  return __builtin_bit_cast(v16bf, t);
}

// B fragment: 32x16 bf16 from an N-major (transposed) buffer: elem(n,k) at n*stride+k.
__device__ __forceinline__ v16bf frag_b(const unsigned short* baseT, int stride,
                                        int colBase, int k0, int lane) {
  const unsigned short* p = baseT + (colBase + (lane & 15)) * stride
                                  + k0 + ((lane >> 4) << 3);
  Pair128 t;
  t.lo = *reinterpret_cast<const u32x4*>(p);
  t.hi = *reinterpret_cast<const u32x4*>(p + 16);
  return __builtin_bit_cast(v16bf, t);
}

__device__ __forceinline__ v8f wmma_bf16(v16bf a, v16bf b, v8f c) {
  return __builtin_amdgcn_wmma_f32_16x16x32_bf16(false, a, false, b,
                                                 (short)0, c, false, false);
}

// ---------- kernel 0: one-time weight conversion ----------
// WtG[h][t][d]  = bf16(transform_weights[h][d][t])        (t-major for B frags)
// B2tG[c][k]    = bf16(k<128 ? neigh_weights[k][c] : self_weights[k-128][c])
__global__ void gat_prep(const float* __restrict__ tw, const float* __restrict__ nw,
                         const float* __restrict__ sw,
                         unsigned short* __restrict__ WtG,
                         unsigned short* __restrict__ B2tG) {
  int i = blockIdx.x * blockDim.x + threadIdx.x;
  int stride = gridDim.x * blockDim.x;
  for (int e = i; e < NHEADS * DIM * DIM; e += stride) {
    int h = e >> 14, rem = e & 16383, t = rem >> 7, d = rem & 127;
    WtG[e] = f32_to_bf16(tw[h * (DIM * DIM) + d * DIM + t]);
  }
  for (int e = i; e < 128 * 256; e += stride) {
    int c = e >> 8, k = e & 255;
    float v = (k < 128) ? nw[k * 128 + c] : sw[(k - 128) * 128 + c];
    B2tG[e] = f32_to_bf16(v);
  }
}

// ---------- kernel 1: fused per-node attention ----------
// LDS layout (bytes):
//   [0,65536)          Wt bf16 [H][T][D]
//   [65536,77824)      A  bf16 [48][128]  (neighbors + self + zero pad)
//   [77824,113728)     nt f32  [H][NT_ROWS][NT_STR]
//   [113728,115776)    aw f32  [512]
//   [115776,116048)    logit f32 [H][34]
//   [116048,116320)    attn  f32 [H][32] (padded)
//   [116320,150112)    stage f32 [2][33*128]  (async double buffer)
#define OFF_A     65536
#define OFF_NT    77824
#define OFF_AW    113728
#define OFF_LOG   115776
#define OFF_ATT   116048
#define OFF_STG   116320
#define SMEM1     150144

__global__ void __launch_bounds__(256, 1)
gat_attention(const float* __restrict__ self_vecs,
              const float* __restrict__ neigh_vecs,
              const float* __restrict__ attn_w,
              const unsigned short* __restrict__ WtG,
              unsigned short* __restrict__ A2, int N) {
  extern __shared__ char smem[];
  unsigned short* lds_wt = (unsigned short*)smem;
  unsigned short* lds_a  = (unsigned short*)(smem + OFF_A);
  float* lds_nt    = (float*)(smem + OFF_NT);
  float* lds_aw    = (float*)(smem + OFF_AW);
  float* lds_logit = (float*)(smem + OFF_LOG);
  float* lds_attn  = (float*)(smem + OFF_ATT);
  float* lds_stg   = (float*)(smem + OFF_STG);   // [2][33*128]

  const int tid  = threadIdx.x;
  const int lane = tid & 31;
  const int w    = tid >> 5;
  const int h    = w >> 2;            // head handled by this wave
  const int colS = (w & 3) * 32;      // 32-column strip within T

  // stage transform weights into LDS (64KB bf16, once)
  {
    const u32x4* src = reinterpret_cast<const u32x4*>(WtG);
    u32x4*       dst = reinterpret_cast<u32x4*>(lds_wt);
    for (int g = tid; g < (NHEADS * DIM * DIM) / 8; g += 256) dst[g] = src[g];
  }
  for (int i = tid; i < 512; i += 256) lds_aw[i] = attn_w[i];
  for (int i = tid; i < (AROWS - 33) * DIM; i += 256) lds_a[33 * DIM + i] = 0;

  // prologue: async-stage node blockIdx.x's activations into buffer 0
  {
    int n0 = blockIdx.x;
    if (n0 < N) {
      const float* nsrc = neigh_vecs + (size_t)n0 * (NS * DIM);
      const float* ssrc = self_vecs + (size_t)n0 * DIM;
      for (int g = tid; g < 33 * 32; g += 256) {       // 33 rows * 32 x 16B chunks
        int row = g >> 5, c4 = (g & 31) << 2;
        const float* src = (row < 32) ? (nsrc + row * DIM + c4) : (ssrc + c4);
        async_g2l_b128(lds_off(lds_stg + row * DIM + c4), src);
      }
    }
  }

  int cur = 0;
  for (int n = blockIdx.x; n < N; n += gridDim.x) {
    // ---- wait for current staged block; publish to all waves ----
    wait_async0();
    __syncthreads();

    // ---- kick off async stage of the NEXT node into the other buffer ----
    int nn = n + gridDim.x;
    if (nn < N) {
      float* stg = lds_stg + (cur ^ 1) * (33 * DIM);
      const float* nsrc = neigh_vecs + (size_t)nn * (NS * DIM);
      const float* ssrc = self_vecs + (size_t)nn * DIM;
      for (int g = tid; g < 33 * 32; g += 256) {
        int row = g >> 5, c4 = (g & 31) << 2;
        const float* src = (row < 32) ? (nsrc + row * DIM + c4) : (ssrc + c4);
        async_g2l_b128(lds_off(stg + row * DIM + c4), src);
      }
    }

    // ---- convert staged fp32 -> bf16 A matrix (LDS -> LDS) ----
    {
      const float* stg = lds_stg + cur * (33 * DIM);
      for (int g = tid; g < 33 * 32; g += 256) {
        int row = g >> 5, c4 = (g & 31) << 2;
        f32x4 v = *reinterpret_cast<const f32x4*>(stg + row * DIM + c4);
        unsigned short* d = lds_a + row * DIM + c4;
        d[0] = f32_to_bf16(v.x); d[1] = f32_to_bf16(v.y);
        d[2] = f32_to_bf16(v.z); d[3] = f32_to_bf16(v.w);
      }
    }
    __syncthreads();

    // ---- WMMA: per wave 3 row tiles x 2 col tiles, K = 128 ----
    {
      v8f acc[3][2];
#pragma unroll
      for (int r = 0; r < 3; ++r)
#pragma unroll
        for (int c = 0; c < 2; ++c) acc[r][c] = {};
      const unsigned short* wt = lds_wt + h * (DIM * DIM);
#pragma unroll
      for (int kb = 0; kb < 128; kb += 32) {
        v16bf aF[3], bF[2];
#pragma unroll
        for (int r = 0; r < 3; ++r) aF[r] = frag_a(lds_a, DIM, r * 16, kb, lane);
#pragma unroll
        for (int c = 0; c < 2; ++c) bF[c] = frag_b(wt, DIM, colS + c * 16, kb, lane);
#pragma unroll
        for (int r = 0; r < 3; ++r)
#pragma unroll
          for (int c = 0; c < 2; ++c) acc[r][c] = wmma_bf16(aF[r], bF[c], acc[r][c]);
      }
      // spill only rows 0..32 of neigh_t/self_t (rest is zero padding)
      float* nt = lds_nt + h * (NT_ROWS * NT_STR);
#pragma unroll
      for (int r = 0; r < 3; ++r) {
        int rowb = r * 16 + ((lane >> 4) << 3);
#pragma unroll
        for (int c = 0; c < 2; ++c) {
          int col = colS + c * 16 + (lane & 15);
#pragma unroll
          for (int i = 0; i < 8; ++i) {
            int row = rowb + i;
            if (row < 33) nt[row * NT_STR + col] = acc[r][c][i];
          }
        }
      }
    }
    __syncthreads();

    // ---- logits: dot(neigh_t[s], a_neigh) / dot(self_t, a_self) ----
    if ((tid & 127) < 33) {
      int hh = tid >> 7, s = tid & 127;
      const float* arow = lds_aw + hh * 256 + ((s == 32) ? 0 : 128);
      const float* ntr  = lds_nt + hh * (NT_ROWS * NT_STR) + s * NT_STR;
      float acc = 0.f;
      for (int k = 0; k < DIM; ++k) acc += ntr[k] * arow[k];
      lds_logit[hh * 34 + s] = acc;
    }
    __syncthreads();

    // ---- leaky-relu + softmax over S=32, one wave per head ----
    if ((w & 3) == 0) {
      float x = lds_logit[h * 34 + 32] + lds_logit[h * 34 + lane];
      x = (x > 0.f) ? x : LEAKY * x;
      float m = x;
#pragma unroll
      for (int o = 16; o > 0; o >>= 1) m = fmaxf(m, __shfl_xor(m, o, 32));
      float e = __expf(x - m);
      float ssum = e;
#pragma unroll
      for (int o = 16; o > 0; o >>= 1) ssum += __shfl_xor(ssum, o, 32);
      lds_attn[h * 32 + lane] = e / ssum;
    }
    __syncthreads();

    // ---- aggregate, mean over heads, elu; emit fused [final_neigh | self] bf16 ----
    if (tid < 128) {
      int t = tid;
      float acc = 0.f;
      for (int hh = 0; hh < NHEADS; ++hh) {
        const float* nt = lds_nt + hh * (NT_ROWS * NT_STR);
        const float* at = lds_attn + hh * 32;
        float a0 = 0.f;
        for (int s = 0; s < NS; ++s) a0 += at[s] * nt[s * NT_STR + t];
        acc += a0;
      }
      float v = acc * 0.5f;                        // mean over H=2
      v = (v > 0.f) ? v : (__expf(v) - 1.f);       // elu
      A2[(size_t)n * 256 + t] = f32_to_bf16(v);
    } else {
      int t = tid - 128;
      A2[(size_t)n * 256 + 128 + t] = f32_to_bf16(self_vecs[(size_t)n * DIM + t]);
    }
    __syncthreads();
    cur ^= 1;
  }
}

// ---------- kernel 2: out = relu([fn|self] (Nx256) @ [Wn;Ws] (256x128)) ----------
#define SMEM2 (32768 + 65536)
__global__ void __launch_bounds__(256, 1)
gat_output(const unsigned short* __restrict__ A2,
           const unsigned short* __restrict__ B2tG,
           float* __restrict__ out, int N) {
  extern __shared__ char smem[];
  unsigned short* lds_A = (unsigned short*)smem;           // [64][256] bf16
  unsigned short* lds_B = (unsigned short*)(smem + 32768); // [128][256] bf16 (N-major)

  const int tid = threadIdx.x, lane = tid & 31, w = tid >> 5;
  const int tileBase = blockIdx.x * 64;

  // async-stage B (entire 256x128, already bf16) and the A tile straight to LDS
  {
    const char* bsrc = (const char*)B2tG;
    for (int g = tid; g < (128 * 256 * 2) / 16; g += 256)
      async_g2l_b128(lds_off((char*)lds_B + g * 16), bsrc + g * 16);
  }
  {
    for (int g = tid; g < (64 * 256 * 2) / 16; g += 256) {
      int node = tileBase + (g >> 5);                 // 32 x 16B chunks per row
      if (node < N) {
        const char* src = (const char*)(A2 + (size_t)node * 256) + (g & 31) * 16;
        async_g2l_b128(lds_off((char*)lds_A + g * 16), src);
      } else {
        u32x4 z = {};
        *reinterpret_cast<u32x4*>((char*)lds_A + g * 16) = z;   // zero-pad tail rows
      }
    }
  }
  wait_async0();
  __syncthreads();

  const int rt = w >> 1;          // row tile 0..3
  const int cs = (w & 1) * 64;    // 64-column strip (4 col tiles)
  v8f acc[4];
#pragma unroll
  for (int c = 0; c < 4; ++c) acc[c] = {};
#pragma unroll
  for (int kb = 0; kb < 256; kb += 32) {
    v16bf aF = frag_a(lds_A, 256, rt * 16, kb, lane);
#pragma unroll
    for (int c = 0; c < 4; ++c) {
      v16bf bF = frag_b(lds_B, 256, cs + c * 16, kb, lane);
      acc[c] = wmma_bf16(aF, bF, acc[c]);
    }
  }
  const int rowb = rt * 16 + ((lane >> 4) << 3);
#pragma unroll
  for (int c = 0; c < 4; ++c) {
    int col = cs + c * 16 + (lane & 15);
#pragma unroll
    for (int i = 0; i < 8; ++i) {
      int node = tileBase + rowb + i;
      if (node < N) out[(size_t)node * 128 + col] = fmaxf(acc[c][i], 0.f);
    }
  }
}

// ---------- launch ----------
extern "C" void kernel_launch(void* const* d_in, const int* in_sizes, int n_in,
                              void* d_out, int out_size, void* d_ws, size_t ws_size,
                              hipStream_t stream) {
  const float* self_vecs  = (const float*)d_in[0];
  const float* neigh_vecs = (const float*)d_in[1];
  const float* tw = (const float*)d_in[2];
  const float* aw = (const float*)d_in[3];
  const float* nw = (const float*)d_in[4];
  const float* sw = (const float*)d_in[5];
  float* out = (float*)d_out;
  const int N = in_sizes[0] / 128;
  if (N <= 0) return;

  char* ws = (char*)d_ws;
  unsigned short* WtG  = (unsigned short*)ws;               // 65536 B
  unsigned short* B2tG = (unsigned short*)(ws + 65536);     // 65536 B
  unsigned short* A2   = (unsigned short*)(ws + 131072);    // N*512 B

  gat_prep<<<64, 256, 0, stream>>>(tw, nw, sw, WtG, B2tG);

  int blocks1 = 2048; if (blocks1 > N) blocks1 = N;
  gat_attention<<<blocks1, 256, SMEM1, stream>>>(self_vecs, neigh_vecs, aw, WtG, A2, N);

  int blocks2 = (N + 63) / 64;
  gat_output<<<blocks2, 256, SMEM2, stream>>>(A2, B2tG, out, N);
}
